// HNNLayer_87385404605202
// MI455X (gfx1250) — matrix operand order repaired and checked by reading, not compiled, not run on hardware
//
#include <hip/hip_runtime.h>
#include <cstdint>

typedef __attribute__((ext_vector_type(16))) _Float16 v16h;
typedef __attribute__((ext_vector_type(8)))  _Float16 v8h;
typedef __attribute__((ext_vector_type(8)))  float    v8f;

#define MIN_NORM 1e-15f
#define MAXN     0.996f   /* 1 - 4e-3 */

__device__ __forceinline__ float artanh_c(float z) {
  z = fminf(fmaxf(z, -1.f + 1e-7f), 1.f - 1e-7f);
  return 0.5f * logf((1.f + z) / (1.f - z));
}
__device__ __forceinline__ float lrelu(float v) { return v > 0.f ? v : 0.01f * v; }
__device__ __forceinline__ float dot4(float4 v) { return v.x*v.x + v.y*v.y + v.z*v.z + v.w*v.w; }

// ---------------------------------------------------------------------------
// Prep: hyp_b = proj(expmap0(b)) for b1 (50 -> padded 64) and b3 (128),
// plus ||hyp_b||^2 scalars.  ws layout: [0..63]=hb1, [64..191]=hb3,
// [192]=Y2_1, [193]=Y2_3.
// ---------------------------------------------------------------------------
__global__ void hnn_prep(const float* __restrict__ b1,
                         const float* __restrict__ b3,
                         float* __restrict__ ws) {
  if (threadIdx.x == 0 && blockIdx.x == 0) {
    {
      float s = 0.f;
      for (int i = 0; i < 50; ++i) s += b1[i] * b1[i];
      float n  = fmaxf(sqrtf(s), MIN_NORM);
      float sc = tanhf(n) / n;
      float vn = tanhf(n);
      float pf = (vn > MAXN) ? (MAXN / vn) : 1.f;
      float y2 = 0.f;
      for (int i = 0; i < 64; ++i) {
        float v = (i < 50) ? sc * pf * b1[i] : 0.f;
        ws[i] = v; y2 += v * v;
      }
      ws[192] = y2;
    }
    {
      float s = 0.f;
      for (int i = 0; i < 128; ++i) s += b3[i] * b3[i];
      float n  = fmaxf(sqrtf(s), MIN_NORM);
      float sc = tanhf(n) / n;
      float vn = tanhf(n);
      float pf = (vn > MAXN) ? (MAXN / vn) : 1.f;
      float y2 = 0.f;
      for (int i = 0; i < 128; ++i) {
        float v = sc * pf * b3[i];
        ws[64 + i] = v; y2 += v * v;
      }
      ws[193] = y2;
    }
  }
}

// ---------------------------------------------------------------------------
// Fused HNN layer: 128 rows per block, 256 threads (8 wave32).
// ---------------------------------------------------------------------------
#define SMEM_BYTES 132872

__global__ __launch_bounds__(256, 1)
void hnn_fused(const float* __restrict__ x, const float* __restrict__ W1,
               const float* __restrict__ W3, const float* __restrict__ ws,
               float* __restrict__ out, int N) {
  extern __shared__ char smem[];
  _Float16* W1H = (_Float16*)(smem);            // 64x512 f16  (64 KB) [aliased by G2 later]
  _Float16* W3H = (_Float16*)(smem + 65536);    // 128x64 f16  (16 KB)
  float*    GT  = (float*)(smem + 81920);       // 128x64 f32  (32 KB)  GEMM1 accum tile
  _Float16* H2H = (_Float16*)(smem + 114688);   // 128x64 f16  (16 KB)  hidden acts
  float*    ROWA = (float*)(smem + 131072);     // 128 f32: tanh(n)/n
  float*    ROWB = (float*)(smem + 131584);     // 128 f32: row norm scalar
  float*    HB1  = (float*)(smem + 132096);     // 64 f32
  float*    HB3  = (float*)(smem + 132352);     // 128 f32
  float*    YS   = (float*)(smem + 132864);     // 2 f32
  float*    G2   = (float*)(smem);              // 128x128 f32 (64 KB) aliases W1H

  const int tid = threadIdx.x;

  // ---- Phase 1: stage weights (f32 -> f16, zero-padded) + constants ----
  for (int i = tid; i < 64 * 512; i += 256) {
    int r = i >> 9, c = i & 511;
    W1H[i] = (_Float16)((r < 50) ? W1[r * 512 + c] : 0.f);
  }
  for (int i = tid; i < 128 * 64; i += 256) {
    int r = i >> 6, c = i & 63;
    W3H[i] = (_Float16)((c < 50) ? W3[r * 50 + c] : 0.f);
  }
  if (tid < 64)  HB1[tid] = ws[tid];
  if (tid < 128) HB3[tid] = ws[64 + tid];
  if (tid < 2)   YS[tid]  = ws[192 + tid];
  __syncthreads();

  const int wave = tid >> 5, lane = tid & 31;
  const int lrow = lane & 15, lhi = lane >> 4;
  const int row0 = blockIdx.x * 128;

  // ---- Phase 2: GEMM1  g = x @ W1^T   (M=16/wave, N=64 padded, K=512) ----
  {
    int grow = row0 + wave * 16 + lrow;
    if (grow >= N) grow = N - 1;
    const float* xr = x + (size_t)grow * 512;
    v8f zero = {};
    v8f acc0 = zero, acc1 = zero, acc2 = zero, acc3 = zero;
    float ss = 0.f;
    for (int ks = 0; ks < 16; ++ks) {
      int kb = ks * 32 + lhi * 16;
      float4 f0 = *(const float4*)(xr + kb);
      float4 f1 = *(const float4*)(xr + kb + 4);
      float4 f2 = *(const float4*)(xr + kb + 8);
      float4 f3 = *(const float4*)(xr + kb + 12);
      ss += dot4(f0) + dot4(f1) + dot4(f2) + dot4(f3);
      v16h a;
      a[0]=(_Float16)f0.x;  a[1]=(_Float16)f0.y;  a[2]=(_Float16)f0.z;  a[3]=(_Float16)f0.w;
      a[4]=(_Float16)f1.x;  a[5]=(_Float16)f1.y;  a[6]=(_Float16)f1.z;  a[7]=(_Float16)f1.w;
      a[8]=(_Float16)f2.x;  a[9]=(_Float16)f2.y;  a[10]=(_Float16)f2.z; a[11]=(_Float16)f2.w;
      a[12]=(_Float16)f3.x; a[13]=(_Float16)f3.y; a[14]=(_Float16)f3.z; a[15]=(_Float16)f3.w;
      int kcol = ks * 32 + lhi * 16;
      v16h b0 = *(const v16h*)(W1H + ((     lrow) * 512 + kcol));
      v16h b1 = *(const v16h*)(W1H + ((16 + lrow) * 512 + kcol));
      v16h b2 = *(const v16h*)(W1H + ((32 + lrow) * 512 + kcol));
      v16h b3 = *(const v16h*)(W1H + ((48 + lrow) * 512 + kcol));
      acc0 = __builtin_amdgcn_wmma_f32_16x16x32_f16(false, a, false, b0, (short)0, acc0, false, false);
      acc1 = __builtin_amdgcn_wmma_f32_16x16x32_f16(false, a, false, b1, (short)0, acc1, false, false);
      acc2 = __builtin_amdgcn_wmma_f32_16x16x32_f16(false, a, false, b2, (short)0, acc2, false, false);
      acc3 = __builtin_amdgcn_wmma_f32_16x16x32_f16(false, a, false, b3, (short)0, acc3, false, false);
    }
    ss += __shfl_xor(ss, 16, 32);        // lanes l and l^16 jointly hold row (l&15)
    if (lane < 16) {
      float n = fmaxf(sqrtf(ss), MIN_NORM);
      ROWA[wave * 16 + lrow] = tanhf(n) / n;              // expmap0 row scale
      ROWB[wave * 16 + lrow] = fmaxf(tanhf(n), MIN_NORM); // ||h|| = tanh(n)
    }
#pragma unroll
    for (int j = 0; j < 8; ++j) {
      int rr = wave * 16 + j + 8 * lhi;
      GT[rr * 64 +      lrow] = acc0[j];
      GT[rr * 64 + 16 + lrow] = acc1[j];
      GT[rr * 64 + 32 + lrow] = acc2[j];
      GT[rr * 64 + 48 + lrow] = acc3[j];
    }
  }
  __syncthreads();

  // ---- Phase 3: row-wise hyp_linear1 epilogue + hyp_act (fp32) ----
  if (tid < 128) {
    float* g = GT + tid * 64;
    float fac = ROWA[tid], xn = ROWB[tid];
    float gn2 = 0.f, dotg = 0.f;
    for (int c = 0; c < 64; ++c) { float v = g[c]; gn2 += v * v; dotg += v * HB1[c]; }
    float gn  = sqrtf(gn2);
    float mxn = fmaxf(fac * gn, MIN_NORM);
    float tt  = (mxn / xn) * artanh_c(xn);
    float sres = tanhf(tt) * fac / mxn;          // res_i = sres * g_i
    if (gn2 == 0.f) sres = 0.f;
    float rn = fmaxf(fabsf(sres) * gn, MIN_NORM);
    if (rn > MAXN) sres *= MAXN / rn;            // proj
    float Y2 = YS[0];
    float xy = sres * dotg, x2 = sres * sres * gn2;
    float ca = 1.f + 2.f * xy + Y2, cb = 1.f - x2;
    float inv = 1.f / fmaxf(1.f + 2.f * xy + x2 * Y2, MIN_NORM);
    float s1 = 0.f, sq = 0.f;
    for (int c = 0; c < 64; ++c) {               // mobius_add, store h1 in place
      float h = (ca * sres * g[c] + cb * HB1[c]) * inv;
      g[c] = h; s1 += h * h;
      float q = lrelu(h); sq += q * q;
    }
    float n1raw = fmaxf(sqrtf(s1), MIN_NORM);
    float ph = (n1raw > MAXN) ? MAXN / n1raw : 1.f;           // proj(h1)
    float n1 = fmaxf(fminf(n1raw, MAXN), MIN_NORM);
    float s1c = artanh_c(n1) / n1;                            // logmap0 scale
    float nq0 = sqrtf(sq);
    float nt = fmaxf(s1c * ph * nq0, MIN_NORM);               // ||xt||
    float th = tanhf(nt);
    float coef = th / nt * s1c * ph;                          // expmap0(lrelu(.))
    float un = fmaxf(fabsf(coef) * nq0, MIN_NORM);
    if (un > MAXN) { coef *= MAXN / un; un = MAXN; }          // proj
    ROWB[tid] = un;                                           // ||h2|| for layer 3
    _Float16* hh = H2H + tid * 64;
    for (int c = 0; c < 64; ++c) hh[c] = (_Float16)(coef * lrelu(g[c]));
  }
  __syncthreads();

  // ---- Phase 4: GEMM2  g2 = h2 @ W3^T  (M=16/wave, N=128, K=64 padded) ----
  {
    v8f zero = {};
    v8f acc[8];
#pragma unroll
    for (int i = 0; i < 8; ++i) acc[i] = zero;
    const _Float16* hrow = H2H + (wave * 16 + lrow) * 64;
    for (int ks = 0; ks < 2; ++ks) {
      int kb = ks * 32 + lhi * 16;
      v8h lo = *(const v8h*)(hrow + kb);
      v8h hi = *(const v8h*)(hrow + kb + 8);
      v16h a = __builtin_shufflevector(lo, hi, 0,1,2,3,4,5,6,7,8,9,10,11,12,13,14,15);
      int kcol = ks * 32 + lhi * 16;
#pragma unroll
      for (int nt = 0; nt < 8; ++nt) {
        v16h b = *(const v16h*)(W3H + (nt * 16 + lrow) * 64 + kcol);
        acc[nt] = __builtin_amdgcn_wmma_f32_16x16x32_f16(false, a, false, b, (short)0, acc[nt], false, false);
      }
    }
#pragma unroll
    for (int nt = 0; nt < 8; ++nt)
#pragma unroll
      for (int j = 0; j < 8; ++j)
        G2[(wave * 16 + j + 8 * lhi) * 128 + nt * 16 + lrow] = acc[nt][j];
  }
  __syncthreads();

  // ---- Phase 5: row-wise hyp_linear3 epilogue + store ----
  if (tid < 128) {
    int grow = row0 + tid;
    float* g = G2 + tid * 128;
    float xn2 = ROWB[tid];
    float gn2 = 0.f, dotg = 0.f;
    for (int c = 0; c < 128; ++c) { float v = g[c]; gn2 += v * v; dotg += v * HB3[c]; }
    float gn  = sqrtf(gn2);
    float mxn = fmaxf(gn, MIN_NORM);
    float tt  = (mxn / xn2) * artanh_c(xn2);
    float sres = tanhf(tt) / mxn;
    if (gn2 == 0.f) sres = 0.f;
    float rn = fmaxf(fabsf(sres) * gn, MIN_NORM);
    if (rn > MAXN) sres *= MAXN / rn;
    float Y2 = YS[1];
    float xy = sres * dotg, x2 = sres * sres * gn2;
    float ca = 1.f + 2.f * xy + Y2, cb = 1.f - x2;
    float inv = 1.f / fmaxf(1.f + 2.f * xy + x2 * Y2, MIN_NORM);
    float s = 0.f;
    for (int c = 0; c < 128; ++c) {
      float h = (ca * sres * g[c] + cb * HB3[c]) * inv;
      g[c] = h; s += h * h;
    }
    float nr = fmaxf(sqrtf(s), MIN_NORM);
    float pf = (nr > MAXN) ? MAXN / nr : 1.f;
    if (grow < N) {
      float4* o = (float4*)(out + (size_t)grow * 128);
      for (int c = 0; c < 32; ++c) {
        float4 v;
        v.x = pf * g[4 * c + 0]; v.y = pf * g[4 * c + 1];
        v.z = pf * g[4 * c + 2]; v.w = pf * g[4 * c + 3];
        o[c] = v;
      }
    }
  }
}

// ---------------------------------------------------------------------------
extern "C" void kernel_launch(void* const* d_in, const int* in_sizes, int n_in,
                              void* d_out, int out_size, void* d_ws, size_t ws_size,
                              hipStream_t stream) {
  const float* x  = (const float*)d_in[0];
  const float* W1 = (const float*)d_in[1];
  const float* b1 = (const float*)d_in[2];
  const float* W3 = (const float*)d_in[3];
  const float* b3 = (const float*)d_in[4];
  float* out = (float*)d_out;
  float* ws  = (float*)d_ws;
  int N = in_sizes[0] / 512;

  hnn_prep<<<1, 32, 0, stream>>>(b1, b3, ws);
  int blocks = (N + 127) / 128;
  hnn_fused<<<blocks, 256, SMEM_BYTES, stream>>>(x, W1, W3, ws, out, N);
}
